// TheoreticalKANLayer_46626164965674
// MI455X (gfx1250) — matrix operand order repaired and checked by reading, not compiled, not run on hardware
//
#include <hip/hip_runtime.h>
#include <hip/hip_bf16.h>

typedef __attribute__((ext_vector_type(16))) _Float16 v16h;
typedef __attribute__((ext_vector_type(8)))  _Float16 v8h;
typedef __attribute__((ext_vector_type(8)))  float    v8f;

#define B_ROWS   32768
#define I_DIM    128
#define O_DIM    256
#define KB       21
#define M_TILE   32
#define A_STRIDE 136    // halves per s_a row: 272B = 16B-aligned, +4 banks/row rotation
#define OUT_STRIDE 260  // floats per s_out row (de-conflict column reads)
#define PI_F 3.14159265358979f

// ---------- order-preserving float <-> u32 keys for deterministic min/max ----------
__device__ __forceinline__ unsigned f2key(float f) {
    unsigned b = __float_as_uint(f);
    return b ^ ((unsigned)((int)b >> 31) | 0x80000000u);
}
__device__ __forceinline__ float key2f(unsigned k) {
    unsigned m = ((int)k >= 0) ? 0xFFFFFFFFu : 0x80000000u;
    return __uint_as_float(k ^ m);
}

__global__ void init_keys_kernel(unsigned* __restrict__ keys) {
    int t = threadIdx.x;
    if (t < 128) keys[t] = 0xFFFFFFFFu;   // min keys
    else         keys[t] = 0u;            // max keys
}

__global__ void colminmax_kernel(const float* __restrict__ x, unsigned* __restrict__ keys) {
    int col = threadIdx.x & 127;
    int rstart = blockIdx.x * 2 + (threadIdx.x >> 7);
    float mn = 3.4e38f, mx = -3.4e38f;
    for (int r = rstart; r < B_ROWS; r += 256) {
        float v = x[(size_t)r * I_DIM + col];
        mn = fminf(mn, v);
        mx = fmaxf(mx, v);
    }
    atomicMin(&keys[col], f2key(mn));
    atomicMax(&keys[128 + col], f2key(mx));
}

// W[kb][o][i] (f16), so a WMMA B-fragment per lane = 16 contiguous halves (32 B)
__global__ void fuse_weights_kernel(const float* __restrict__ bw,
                                    const float* __restrict__ conn,
                                    _Float16* __restrict__ w) {
    int t = blockIdx.x * 256 + threadIdx.x;   // t = o*128 + i, 32768 threads
    float c = conn[t];
    const float* src = bw + (size_t)t * KB;
    #pragma unroll
    for (int kb = 0; kb < KB; ++kb)
        w[(size_t)kb * (O_DIM * I_DIM) + t] = (_Float16)(src[kb] * c);
}

__device__ __forceinline__ float basis_val(int kb, float x) {
    if (kb == 0) return 1.0f;
    if (kb <= 6) {  // Chebyshev recurrence
        float tp = 1.0f, tc = x;
        for (int i = 1; i < kb; ++i) {
            float tn = 2.0f * x * tc - tp;
            tp = tc; tc = tn;
        }
        return tc;
    }
    if (kb <= 14) { // interleaved sin/cos, k = 1..4
        int j = kb - 7;
        float ang = (float)((j >> 1) + 1) * PI_F * x;
        return (j & 1) ? __cosf(ang) : __sinf(ang);
    }
    // RBF, centers linspace(-1,1,6), 1/(2*0.3^2) = 5.5555553
    float cc = -1.0f + 0.4f * (float)(kb - 15);
    float d = x - cc;
    return __expf(d * d * -5.5555553f);
}

__launch_bounds__(256)
__global__ void kan_gemm_kernel(const float* __restrict__ x,
                                const _Float16* __restrict__ w,
                                const unsigned* __restrict__ keys,
                                const float* __restrict__ bias,
                                const float* __restrict__ gamma,
                                const float* __restrict__ beta,
                                float* __restrict__ out) {
    __shared__ float s_scale[128], s_off[128];
    __shared__ float s_bias[O_DIM], s_gamma[O_DIM], s_beta[O_DIM];
    __shared__ float s_mean[M_TILE], s_rstd[M_TILE];
    // union region: (xn f32[32][128] + a f16[32][A_STRIDE]) overlapped with out f32[32][OUT_STRIDE]
    __shared__ __align__(16) unsigned char s_u[M_TILE * OUT_STRIDE * 4];
    float*    s_xn  = (float*)s_u;                               // 16384 B
    _Float16* s_a   = (_Float16*)(s_u + M_TILE * I_DIM * 4);     //  8704 B
    float*    s_out = (float*)s_u;                               // 33280 B

    const int tid  = threadIdx.x;
    const int lane = tid & 31;
    const int wave = tid >> 5;
    const int m0   = blockIdx.x * M_TILE;

    if (tid < 128) {
        float mn = key2f(keys[tid]);
        float mx = key2f(keys[128 + tid]);
        float sc = 2.0f / (mx - mn + 1e-8f);
        s_scale[tid] = sc;
        s_off[tid]   = -sc * mn - 1.0f;
    }
    s_bias[tid]  = bias[tid];
    s_gamma[tid] = gamma[tid];
    s_beta[tid]  = beta[tid];
    __syncthreads();

    // stage normalized x tile (float4 coalesced)
    for (int e = tid; e < M_TILE * I_DIM / 4; e += 256) {
        int r = e >> 5, c4 = (e & 31) * 4;
        float4 xv = *(const float4*)(x + (size_t)(m0 + r) * I_DIM + c4);
        s_xn[r * I_DIM + c4 + 0] = xv.x * s_scale[c4 + 0] + s_off[c4 + 0];
        s_xn[r * I_DIM + c4 + 1] = xv.y * s_scale[c4 + 1] + s_off[c4 + 1];
        s_xn[r * I_DIM + c4 + 2] = xv.z * s_scale[c4 + 2] + s_off[c4 + 2];
        s_xn[r * I_DIM + c4 + 3] = xv.w * s_scale[c4 + 3] + s_off[c4 + 3];
    }
    __syncthreads();

    v8f zero = {0.f, 0.f, 0.f, 0.f, 0.f, 0.f, 0.f, 0.f};
    v8f acc[4];  // [mblk 0..1][nblk 0..1], wave owns cols [wave*32, wave*32+32)
    acc[0] = zero; acc[1] = zero; acc[2] = zero; acc[3] = zero;

    const int o_w    = wave * 32;
    const int a_m    = lane & 15;
    const int a_kh   = (lane >> 4) * 16;   // A: K base (0 or 16)
    const int b_n    = lane & 15;
    const int b_kb16 = (lane >> 4) * 16;   // B: K base (0 or 16)

    for (int kb = 0; kb < KB; ++kb) {
        // regenerate basis plane kb into f16 A-tile
        for (int e = tid; e < M_TILE * I_DIM; e += 256) {
            int r = e >> 7, c = e & 127;
            s_a[r * A_STRIDE + c] = (_Float16)basis_val(kb, s_xn[r * I_DIM + c]);
        }
        __syncthreads();

        const _Float16* wkb = w + (size_t)kb * (O_DIM * I_DIM);
        #pragma unroll
        for (int ks = 0; ks < 4; ++ks) {
            const int kk0 = ks * 32;
            v16h a[2], b[2];
            #pragma unroll
            for (int mb = 0; mb < 2; ++mb) {
                const _Float16* ar = s_a + (mb * 16 + a_m) * A_STRIDE + kk0 + a_kh;
                v8h lo = *(const v8h*)ar;
                v8h hi = *(const v8h*)(ar + 8);
                a[mb] = __builtin_shufflevector(lo, hi,
                        0,1,2,3,4,5,6,7,8,9,10,11,12,13,14,15);
            }
            #pragma unroll
            for (int nb = 0; nb < 2; ++nb) {
                const _Float16* br = wkb + (size_t)(o_w + nb * 16 + b_n) * I_DIM
                                         + kk0 + b_kb16;
                b[nb] = *(const v16h*)br;   // 32 B contiguous -> 2x b128
            }
            #pragma unroll
            for (int mb = 0; mb < 2; ++mb)
                #pragma unroll
                for (int nb = 0; nb < 2; ++nb)
                    acc[mb * 2 + nb] = __builtin_amdgcn_wmma_f32_16x16x32_f16(
                        false, a[mb], false, b[nb],
                        (short)0, acc[mb * 2 + nb], false, false);
        }
        __syncthreads();   // protect s_a before next kb overwrite (and s_out aliasing)
    }

    // spill accumulators (+bias) into LDS out-tile (C/D layout: vgpr r -> M, lane -> N)
    #pragma unroll
    for (int mb = 0; mb < 2; ++mb)
        #pragma unroll
        for (int nb = 0; nb < 2; ++nb) {
            int o     = o_w + nb * 16 + (lane & 15);
            int mbase = mb * 16 + (lane >> 4) * 8;
            v8f a = acc[mb * 2 + nb];
            #pragma unroll
            for (int r = 0; r < 8; ++r)
                s_out[(mbase + r) * OUT_STRIDE + o] = a[r] + s_bias[o];
        }
    __syncthreads();

    // LayerNorm stats: 8 lanes per row, shfl_xor reduce inside wave32
    {
        int r  = tid >> 3;
        int o0 = tid & 7;
        float s = 0.f, s2 = 0.f;
        for (int j = 0; j < 32; ++j) {
            float v = s_out[r * OUT_STRIDE + o0 + j * 8];
            s += v; s2 += v * v;
        }
        #pragma unroll
        for (int d = 1; d < 8; d <<= 1) {
            s  += __shfl_xor(s,  d, 32);
            s2 += __shfl_xor(s2, d, 32);
        }
        if (o0 == 0) {
            float mean = s * (1.0f / 256.0f);
            float var  = s2 * (1.0f / 256.0f) - mean * mean;
            s_mean[r] = mean;
            s_rstd[r] = rsqrtf(var + 1e-5f);
        }
    }
    __syncthreads();

    // normalized, affine, coalesced global store
    for (int it = 0; it < M_TILE; ++it) {
        float v = (s_out[it * OUT_STRIDE + tid] - s_mean[it]) * s_rstd[it]
                  * s_gamma[tid] + s_beta[tid];
        out[(size_t)(m0 + it) * O_DIM + tid] = v;
    }
}

extern "C" void kernel_launch(void* const* d_in, const int* in_sizes, int n_in,
                              void* d_out, int out_size, void* d_ws, size_t ws_size,
                              hipStream_t stream) {
    const float* x     = (const float*)d_in[0];
    const float* bw    = (const float*)d_in[1];
    const float* conn  = (const float*)d_in[2];
    const float* bias  = (const float*)d_in[3];
    const float* gamma = (const float*)d_in[4];
    const float* beta  = (const float*)d_in[5];

    // ws layout: [0,1024): u32 min/max keys; [1024, 1024 + 21*256*128*2): f16 W
    unsigned*  keys = (unsigned*)d_ws;
    _Float16*  wf   = (_Float16*)((char*)d_ws + 1024);

    init_keys_kernel<<<1, 256, 0, stream>>>(keys);
    colminmax_kernel<<<128, 256, 0, stream>>>(x, keys);
    fuse_weights_kernel<<<128, 256, 0, stream>>>(bw, conn, wf);
    kan_gemm_kernel<<<B_ROWS / M_TILE, 256, 0, stream>>>(
        x, wf, keys, bias, gamma, beta, (float*)d_out);
}
